// RelativeMultiHeadAttention_31619549233727
// MI455X (gfx1250) — matrix operand order repaired and checked by reading, not compiled, not run on hardware
//
#include <hip/hip_runtime.h>

// Problem constants (fixed by the reference)
#define SEQ 2048
#define DM  1024
#define NH  16
#define DHD 64

typedef __attribute__((ext_vector_type(16))) __bf16 v16bf;
typedef __attribute__((ext_vector_type(8)))  float  v8f;

union V16 { v16bf v; uint4 q[2]; };

static __device__ inline v8f wmma_bf16(v16bf a, v16bf b, v8f c) {
  // D = A(16x32 bf16) * B(32x16 bf16) + C(16x16 f32)
  return __builtin_amdgcn_wmma_f32_16x16x32_bf16(false, a, false, b, (short)0, c,
                                                 false, false);
}

// CDNA5 async global->LDS copy (ASYNCcnt-tracked, no VGPR round trip).
// LDS byte address = low 32 bits of the generic pointer (LDS aperture rule,
// ISA 10.2). Inline asm per CDNA5 ISA.
static __device__ inline void async_load_b128(const void* g, void* lds) {
  unsigned long long ga = (unsigned long long)(uintptr_t)g;
  unsigned int       la = (unsigned int)(uintptr_t)lds;
  asm volatile("global_load_async_to_lds_b128 %0, %1, off"
               :: "v"(la), "v"(ga) : "memory");
}
static __device__ inline void wait_async0() {
  asm volatile("s_wait_asynccnt 0" ::: "memory");
}

// ---------------------------------------------------------------------------
// Tiled GEMM:  Out[M,N] = A[M,K] @ W[N,K]^T (+ bias0[n] + bias1[n])
// Block = 256 threads (8 waves), C tile 128x64, K staged 32 deep in LDS (bf16).
// A_BF16: A is bf16 (else f32, converted while staging).
// OUT_F32: write f32 (final output), else bf16.
// OUT_TR : write Out transposed as [B,H,DH,S] (used for V so attn*V B-frags
//          are contiguous in the contraction dim).
// ---------------------------------------------------------------------------
template<bool A_BF16, bool OUT_F32, bool OUT_TR>
__global__ __launch_bounds__(256) void gemm_wmma_kernel(
    const void* __restrict__ A_, const float* __restrict__ W,
    const float* __restrict__ bias0, const float* __restrict__ bias1,
    void* __restrict__ Out_, int M, int N, int K)
{
  (void)M;
  __shared__ __bf16 As[128 * 32];   // 8 KB
  __shared__ __bf16 Bs[64 * 32];    // 4 KB

  const int tileM = blockIdx.y * 128;
  const int tileN = blockIdx.x * 64;
  const int tid   = threadIdx.x;
  const int w     = tid >> 5;
  const int lane  = tid & 31;
  const int half  = lane >> 4;
  const int lj    = lane & 15;

  const v8f zero = {0.f,0.f,0.f,0.f,0.f,0.f,0.f,0.f};
  v8f acc[4];
#pragma unroll
  for (int t = 0; t < 4; ++t) acc[t] = zero;

  for (int k0 = 0; k0 < K; k0 += 32) {
    // ---- stage A tile: 128 rows x 32 K (convert f32->bf16 if needed) ----
    {
      const int row = tid >> 1;
      const int seg = (tid & 1) * 16;
      if (A_BF16) {
        const __bf16* ap = (const __bf16*)A_ + (size_t)(tileM + row) * K + k0 + seg;
        *(uint4*)&As[row * 32 + seg]     = *(const uint4*)ap;
        *(uint4*)&As[row * 32 + seg + 8] = *(const uint4*)(ap + 8);
      } else {
        const float* ap = (const float*)A_ + (size_t)(tileM + row) * K + k0 + seg;
#pragma unroll
        for (int j = 0; j < 4; ++j) {
          float4 f = ((const float4*)ap)[j];
          As[row * 32 + seg + j * 4 + 0] = (__bf16)f.x;
          As[row * 32 + seg + j * 4 + 1] = (__bf16)f.y;
          As[row * 32 + seg + j * 4 + 2] = (__bf16)f.z;
          As[row * 32 + seg + j * 4 + 3] = (__bf16)f.w;
        }
      }
    }
    // ---- stage B tile: 64 rows (of W) x 32 K ----
    {
      const int row = tid >> 2;
      const int seg = (tid & 3) * 8;
      const float* wp = W + (size_t)(tileN + row) * K + k0 + seg;
#pragma unroll
      for (int j = 0; j < 2; ++j) {
        float4 f = ((const float4*)wp)[j];
        Bs[row * 32 + seg + j * 4 + 0] = (__bf16)f.x;
        Bs[row * 32 + seg + j * 4 + 1] = (__bf16)f.y;
        Bs[row * 32 + seg + j * 4 + 2] = (__bf16)f.z;
        Bs[row * 32 + seg + j * 4 + 3] = (__bf16)f.w;
      }
    }
    __syncthreads();

    // A fragment for this wave's 16 rows (16x32, A-layout chunks)
    V16 a;
    const __bf16* ar = &As[(w * 16 + lj) * 32];
    a.q[0] = *(const uint4*)(ar + half * 16);
    a.q[1] = *(const uint4*)(ar + half * 16 + 8);

#pragma unroll
    for (int t = 0; t < 4; ++t) {
      V16 bf;
      const __bf16* br = &Bs[(t * 16 + lj) * 32 + half * 16];
      bf.q[0] = ((const uint4*)br)[0];
      bf.q[1] = ((const uint4*)br)[1];
      acc[t] = wmma_bf16(a.v, bf.v, acc[t]);
    }
    __syncthreads();
  }

  // ---- epilogue: bias add, convert, store ----
#pragma unroll
  for (int t = 0; t < 4; ++t) {
    const int gcol = tileN + t * 16 + lj;
    float badd = 0.f;
    if (bias0) badd += bias0[gcol];
    if (bias1) badd += bias1[gcol];
#pragma unroll
    for (int r = 0; r < 8; ++r) {
      const int grow = tileM + w * 16 + r + 8 * half;
      const float val = acc[t][r] + badd;
      if (OUT_F32) {
        ((float*)Out_)[(size_t)grow * N + gcol] = val;
      } else if (OUT_TR) {
        const int bb = grow >> 11;        // / SEQ
        const int s  = grow & (SEQ - 1);
        const int hh = gcol >> 6;         // / DHD
        const int dh = gcol & (DHD - 1);
        ((__bf16*)Out_)[(((size_t)(bb * NH + hh) * DHD + dh) << 11) + s] = (__bf16)val;
      } else {
        ((__bf16*)Out_)[(size_t)grow * N + gcol] = (__bf16)val;
      }
    }
  }
}

// ---------------------------------------------------------------------------
// Fused flash attention with Transformer-XL relative shift.
// Grid: (B*NH, SEQ/128). Block: 256 threads (8 waves); each wave owns a 16-row
// Q tile, all waves share one (b,h) so K/V tiles are staged in LDS once per
// block via async global->LDS copies (ASYNCcnt), overlapped with the position
// term (PH WMMAs + ds_bpermute diagonal gather) which reads global directly.
// ---------------------------------------------------------------------------
__global__ __launch_bounds__(256) void relattn_kernel(
    const __bf16* __restrict__ QU, const __bf16* __restrict__ QV,
    const __bf16* __restrict__ KH, const __bf16* __restrict__ PH,
    const __bf16* __restrict__ VHT, __bf16* __restrict__ AO)
{
  __shared__ __bf16 Ks[32 * 64];       // K tile  [kv][dh], 4 KB
  __shared__ __bf16 Vs[64 * 32];       // V^T tile [dh][kv], 4 KB
  __shared__ __bf16 Pt[8 * 16 * 32];   // per-wave P transpose buffer, 8 KB

  const int bh   = blockIdx.x;
  const int b    = bh >> 4;            // / NH
  const int h    = bh & (NH - 1);
  const int tid  = threadIdx.x;
  const int w    = tid >> 5;
  const int lane = tid & 31;
  const int half = lane >> 4;
  const int lj   = lane & 15;
  const int i0   = blockIdx.y * 128 + w * 16;
  __bf16* wp_ = &Pt[w * (16 * 32)];

  // Per-thread staging coordinates (256 threads cover both tiles)
  const int krow = tid >> 3, kseg = (tid & 7) * 8;   // Ks: 32 x 64
  const int vrow = tid >> 2, vseg = (tid & 3) * 8;   // Vs: 64 x 32
  const __bf16* ksrc0 = KH  + ((size_t)(b * SEQ + krow)) * DM + h * DHD + kseg;
  const __bf16* vsrc0 = VHT + ((size_t)((b * NH + h) * DHD + vrow)) * SEQ + vseg;

  // Preload Q fragments (u-biased and v-biased), 2 K-steps over DH=64.
  const __bf16* qurow = QU + ((size_t)(b * SEQ + i0 + lj)) * DM + h * DHD;
  const __bf16* qvrow = QV + ((size_t)(b * SEQ + i0 + lj)) * DM + h * DHD;
  V16 aqu[2], aqv[2];
#pragma unroll
  for (int f = 0; f < 2; ++f) {
    aqu[f].q[0] = *(const uint4*)(qurow + f * 32 + half * 16);
    aqu[f].q[1] = *(const uint4*)(qurow + f * 32 + half * 16 + 8);
    aqv[f].q[0] = *(const uint4*)(qvrow + f * 32 + half * 16);
    aqv[f].q[1] = *(const uint4*)(qvrow + f * 32 + half * 16 + 8);
  }

  const v8f zero = {0.f,0.f,0.f,0.f,0.f,0.f,0.f,0.f};
  v8f o[4] = {zero, zero, zero, zero};
  float mrow[8], lrow[8];
#pragma unroll
  for (int r = 0; r < 8; ++r) { mrow[r] = -3.0e38f; lrow[r] = 0.f; }

  for (int j0 = 0; j0 < SEQ; j0 += 32) {
    // All waves finished consuming the previous K/V tiles.
    __syncthreads();

    // Issue async K/V tile copies into LDS (tracked by ASYNCcnt).
    async_load_b128(ksrc0 + (size_t)j0 * DM, &Ks[krow * 64 + kseg]);
    async_load_b128(vsrc0 + j0,              &Vs[vrow * 32 + vseg]);
    if (j0 + 32 < SEQ) {
      __builtin_prefetch(ksrc0 + (size_t)(j0 + 32) * DM, 0, 1);  // global_prefetch_b8
    }

    // ---- position term (global PH reads) overlaps the async copies ----
    float sp[2][8];
#pragma unroll
    for (int sub = 0; sub < 2; ++sub) {
      const int js = j0 + sub * 16;
      const int rbase = SEQ - 16 - i0 + js;
      int r0 = rbase + lj;       r0 = r0 < 0 ? 0 : (r0 > SEQ - 1 ? SEQ - 1 : r0);
      int r1 = rbase + 16 + lj;  r1 = r1 < 0 ? 0 : (r1 > SEQ - 1 ? SEQ - 1 : r1);
      const __bf16* prow0 = PH + ((size_t)(b * SEQ + r0)) * DM + h * DHD + half * 16;
      const __bf16* prow1 = PH + ((size_t)(b * SEQ + r1)) * DM + h * DHD + half * 16;
      v8f pa = zero, pb = zero;
      V16 bp;
      bp.q[0] = *(const uint4*)(prow0);      bp.q[1] = *(const uint4*)(prow0 + 8);
      pa = wmma_bf16(aqv[0].v, bp.v, pa);
      bp.q[0] = *(const uint4*)(prow0 + 32); bp.q[1] = *(const uint4*)(prow0 + 40);
      pa = wmma_bf16(aqv[1].v, bp.v, pa);
      bp.q[0] = *(const uint4*)(prow1);      bp.q[1] = *(const uint4*)(prow1 + 8);
      pb = wmma_bf16(aqv[0].v, bp.v, pb);
      bp.q[0] = *(const uint4*)(prow1 + 32); bp.q[1] = *(const uint4*)(prow1 + 40);
      pb = wmma_bf16(aqv[1].v, bp.v, pb);

      // rel-shift diagonal gather: out[li, lj] = P[li, 15 - li + lj]
#pragma unroll
      for (int r = 0; r < 8; ++r) {
        const int li  = r + 8 * half;
        const int c   = 15 - li + lj;              // 0..30, always valid
        const int src = (c & 15) | (lane & 16);
        const float g0 = __shfl(pa[r], src, 32);
        const float g1 = __shfl(pb[r], src, 32);
        sp[sub][r] = (c < 16) ? g0 : g1;
      }
    }

    // K/V tiles ready: wait own async issues, then block barrier.
    wait_async0();
    __syncthreads();

    // ---- content term from the shared K tile in LDS ----
    float ssc[2][8];
#pragma unroll
    for (int sub = 0; sub < 2; ++sub) {
      const __bf16* krl = &Ks[(sub * 16 + lj) * 64 + half * 16];
      V16 b0, b1;
      b0.q[0] = *(const uint4*)(krl);
      b0.q[1] = *(const uint4*)(krl + 8);
      b1.q[0] = *(const uint4*)(krl + 32);
      b1.q[1] = *(const uint4*)(krl + 40);
      v8f cacc = zero;
      cacc = wmma_bf16(aqu[0].v, b0.v, cacc);
      cacc = wmma_bf16(aqu[1].v, b1.v, cacc);
#pragma unroll
      for (int r = 0; r < 8; ++r)
        ssc[sub][r] = (cacc[r] + sp[sub][r]) * 0.03125f;   // / sqrt(D) = /32
    }

    // ---- online softmax over the 32 new columns ----
    v8f scv;
#pragma unroll
    for (int r = 0; r < 8; ++r) {
      const int li = r + 8 * half;
      float mt = fmaxf(ssc[0][r], ssc[1][r]);
      mt = fmaxf(mt, __shfl_xor(mt, 1, 32));
      mt = fmaxf(mt, __shfl_xor(mt, 2, 32));
      mt = fmaxf(mt, __shfl_xor(mt, 4, 32));
      mt = fmaxf(mt, __shfl_xor(mt, 8, 32));
      const float mn = fmaxf(mrow[r], mt);
      const float sc = __expf(mrow[r] - mn);
      mrow[r] = mn;
      const float p0 = __expf(ssc[0][r] - mn);
      const float p1 = __expf(ssc[1][r] - mn);
      wp_[li * 32 + lj]      = (__bf16)p0;   // transpose P through LDS
      wp_[li * 32 + 16 + lj] = (__bf16)p1;
      float rs = p0 + p1;
      rs += __shfl_xor(rs, 1, 32);
      rs += __shfl_xor(rs, 2, 32);
      rs += __shfl_xor(rs, 4, 32);
      rs += __shfl_xor(rs, 8, 32);
      lrow[r] = lrow[r] * sc + rs;
      scv[r] = sc;
    }
#pragma unroll
    for (int t = 0; t < 4; ++t) o[t] *= scv;

    // wave-private LDS RAW: ds ops are in-order per wave; wait for the stores
    asm volatile("s_wait_dscnt 0" ::: "memory");

    // Reload P as a 16x32 A fragment
    V16 ap;
    const __bf16* pr = &wp_[lj * 32];
    ap.q[0] = *(const uint4*)(pr + half * 16);
    ap.q[1] = *(const uint4*)(pr + half * 16 + 8);

    // O += P (16x32) x V (32x64) from the shared V^T tile in LDS
#pragma unroll
    for (int t = 0; t < 4; ++t) {
      const __bf16* vrl = &Vs[(t * 16 + lj) * 32 + half * 16];
      V16 bv_;
      bv_.q[0] = *(const uint4*)(vrl);
      bv_.q[1] = *(const uint4*)(vrl + 8);
      o[t] = wmma_bf16(ap.v, bv_.v, o[t]);
    }
  }

  // ---- normalize and store bf16 [B,S,D] for the final GEMM ----
#pragma unroll
  for (int r = 0; r < 8; ++r) lrow[r] = 1.0f / lrow[r];
#pragma unroll
  for (int t = 0; t < 4; ++t) {
    const int col = h * DHD + t * 16 + lj;
#pragma unroll
    for (int r = 0; r < 8; ++r) {
      const int row = i0 + r + 8 * half;
      AO[((size_t)(b * SEQ + row)) * DM + col] = (__bf16)(o[t][r] * lrow[r]);
    }
  }
}

// ---------------------------------------------------------------------------
// Launcher. Inputs (setup_inputs order):
//  0:q 1:k 2:v 3:pos_embedding 4:Wq 5:bq 6:Wk 7:bk 8:Wv 9:bv 10:Wp
//  11:u_bias 12:v_bias 13:Wo 14:bo
// Workspace (bf16): QU, QV, KH, VHT, PH, AO -- 6 x 4096x1024 = 48 MiB.
// ---------------------------------------------------------------------------
extern "C" void kernel_launch(void* const* d_in, const int* in_sizes, int n_in,
                              void* d_out, int out_size, void* d_ws, size_t ws_size,
                              hipStream_t stream) {
  (void)in_sizes; (void)n_in; (void)out_size; (void)ws_size;
  const float* q  = (const float*)d_in[0];
  const float* k  = (const float*)d_in[1];
  const float* v  = (const float*)d_in[2];
  const float* pe = (const float*)d_in[3];
  const float* Wq = (const float*)d_in[4];
  const float* bq = (const float*)d_in[5];
  const float* Wk = (const float*)d_in[6];
  const float* bk = (const float*)d_in[7];
  const float* Wv = (const float*)d_in[8];
  const float* bv = (const float*)d_in[9];
  const float* Wp = (const float*)d_in[10];
  const float* ub = (const float*)d_in[11];
  const float* vb = (const float*)d_in[12];
  const float* Wo = (const float*)d_in[13];
  const float* bo = (const float*)d_in[14];

  __bf16* ws = (__bf16*)d_ws;
  const size_t T = (size_t)(2 * SEQ) * DM;   // 4096*1024 elements per tensor
  __bf16* QU  = ws + 0 * T;
  __bf16* QV  = ws + 1 * T;
  __bf16* KH  = ws + 2 * T;
  __bf16* VHT = ws + 3 * T;
  __bf16* PH  = ws + 4 * T;
  __bf16* AO  = ws + 5 * T;

  dim3 blk(256);
  dim3 gG(DM / 64, (2 * SEQ) / 128);   // (16, 32) for the 4096x1024 GEMMs
  dim3 gA(2 * NH, SEQ / 128);          // (32, 16) for attention

  // Projections (f32 in, bf16 out). Q is projected twice with u/v head biases
  // folded in; V is written transposed [B,H,DH,S].
  gemm_wmma_kernel<false,false,false><<<gG, blk, 0, stream>>>(q,  Wq, bq, ub, (void*)QU,  4096, DM, DM);
  gemm_wmma_kernel<false,false,false><<<gG, blk, 0, stream>>>(q,  Wq, bq, vb, (void*)QV,  4096, DM, DM);
  gemm_wmma_kernel<false,false,false><<<gG, blk, 0, stream>>>(k,  Wk, bk, nullptr, (void*)KH,  4096, DM, DM);
  gemm_wmma_kernel<false,false,true ><<<gG, blk, 0, stream>>>(v,  Wv, bv, nullptr, (void*)VHT, 4096, DM, DM);
  gemm_wmma_kernel<false,false,false><<<gG, blk, 0, stream>>>(pe, Wp, nullptr, nullptr, (void*)PH, 4096, DM, DM);

  // Fused flash attention with relative shift
  relattn_kernel<<<gA, blk, 0, stream>>>(QU, QV, KH, PH, VHT, AO);

  // Output projection (bf16 in, f32 out to d_out)
  gemm_wmma_kernel<true,true,false><<<gG, blk, 0, stream>>>(AO, Wo, bo, nullptr, d_out, 4096, DM, DM);
}
